// DrugFEM_30279519436889
// MI455X (gfx1250) — compile-verified, matrix operand order checked by reading, not run on hardware
//
#include <hip/hip_runtime.h>
#include <hip/hip_bf16.h>
#include <math.h>
#include <stdint.h>

#define N_NODES 100000
#define N_EDGES 1600000
#define DIM     128
#define NGRAPH  4096
#define CHALF   64
#define ETOTAL  (N_EDGES + N_NODES)
#define GAT_SLOPE_C 0.2f
#define ACT_SLOPE_C 0.01f
#define BN_EPS_C    1e-5f

typedef __attribute__((ext_vector_type(16))) __bf16 v16bf;
typedef __attribute__((ext_vector_type(8)))  __bf16 v8bf;
typedef __attribute__((ext_vector_type(8)))  float  v8f;
typedef int v4i_ __attribute__((vector_size(16)));   // matches async-LDS builtin param

#ifndef __has_builtin
#define __has_builtin(x) 0
#endif

#if defined(__gfx1250__) && __has_builtin(__builtin_amdgcn_global_load_async_to_lds_b128)
#define HAVE_ASYNC_LDS 1
#else
#define HAVE_ASYNC_LDS 0
#endif

static __device__ __forceinline__ __bf16 f2bf(float f) {
  unsigned u = __builtin_bit_cast(unsigned, f);
  unsigned r = (u + 0x7FFFu + ((u >> 16) & 1u)) >> 16;   // round-to-nearest-even
  return __builtin_bit_cast(__bf16, (unsigned short)r);
}

static __device__ __forceinline__ float leakyf(float v, float s) {
  return v > 0.0f ? v : v * s;
}

static __device__ __forceinline__ void atomicMaxF(float* addr, float v) {
  if (v >= 0.0f) atomicMax((int*)addr, __float_as_int(v));
  else           atomicMin((unsigned int*)addr, __float_as_uint(v));
}

// ---------------------------------------------------------------------------
// Generic fill
// ---------------------------------------------------------------------------
__global__ void k_fill(float* p, float v, size_t n) {
  size_t i = (size_t)blockIdx.x * blockDim.x + threadIdx.x;
  size_t stride = (size_t)gridDim.x * blockDim.x;
  for (; i < n; i += stride) p[i] = v;
}

// ---------------------------------------------------------------------------
// Pack B[K x Nn] (row-major fp32) into WMMA B-fragment order, bf16.
// One wave per (column-tile, K-chunk); lane writes its 16 contiguous elements.
// Bp layout: [(tile*nchunks + chunk)*32 + lane][16]
// ---------------------------------------------------------------------------
__global__ void k_pack_b(const float* __restrict__ B, __bf16* __restrict__ Bp,
                         int K, int Nn) {
  const int wid  = blockIdx.x * (blockDim.x >> 5) + (threadIdx.x >> 5);
  const int lane = threadIdx.x & 31;
  const int nchunks = K >> 5;
  const int tiles_n = Nn >> 4;
  if (wid >= tiles_n * nchunks) return;
  const int tn = wid / nchunks, c = wid % nchunks;
  const int col0 = tn << 4, kk = c << 5;
  const int half = lane >> 4, lid = lane & 15;
  __bf16* o = Bp + ((size_t)wid * 32 + lane) * 16;
#pragma unroll
  for (int e = 0; e < 16; ++e) {
    const int k = ((e < 8) ? 0 : 16) + half * 8 + (e & 7);
    o[e] = f2bf(B[(size_t)(kk + k) * Nn + col0 + lid]);
  }
}

// ---------------------------------------------------------------------------
// WMMA GEMM: C[M x Nn] = A[M x K] @ B[K x Nn] (+ bias[Nn]).
// A: row-major fp32; Bp: pre-packed bf16 fragments (k_pack_b).
// Block = 256 threads = 8 wave32. The block stages its 16-row A strip into
// LDS (async global->LDS copy when available), converts once to bf16, then
// wave w computes output-column tile w via v_wmma_f32_16x16x32_bf16 with
// A fragments read as two ds_load_b128 per K-chunk and B fragments as one
// 32B global vector load. Requires M%16==0, K%32==0 (K<=128), Nn%16==0 (<=128).
// ---------------------------------------------------------------------------
__global__ void __launch_bounds__(256)
k_wmma_gemm(const float* __restrict__ A, const __bf16* __restrict__ Bp,
            const float* __restrict__ bias, float* __restrict__ C,
            int M, int K, int Nn) {
  __shared__ __align__(32) float  As[16 * 128];
  __shared__ __align__(32) __bf16 Abf[16 * 128];

  const int t    = threadIdx.x;
  const int lane = t & 31;
  const int wave = t >> 5;
  const int row0 = blockIdx.x << 4;
  const int strip = 16 * K;                 // fp32 elements in this A strip

  // ---- stage A strip into LDS -------------------------------------------
#if HAVE_ASYNC_LDS
  for (int i = t * 4; i < strip; i += 1024) {
    __builtin_amdgcn_global_load_async_to_lds_b128(
        (__attribute__((address_space(1))) v4i_*)(A + (size_t)row0 * K + i),
        (__attribute__((address_space(3))) v4i_*)(&As[i]),
        0, 0);
  }
#if __has_builtin(__builtin_amdgcn_s_wait_asynccnt)
  __builtin_amdgcn_s_wait_asynccnt(0);
#else
  asm volatile("s_wait_asynccnt 0x0" ::: "memory");
#endif
#else
  for (int i = t * 4; i < strip; i += 1024) {
    *(float4*)&As[i] = *(const float4*)(A + (size_t)row0 * K + i);
  }
#endif
  __syncthreads();

  // ---- one-shot fp32 -> bf16 conversion in LDS --------------------------
  for (int i = t; i < strip; i += 256) Abf[i] = f2bf(As[i]);
  __syncthreads();

  // ---- per-wave 16x16 output tile ---------------------------------------
  const int tiles_n = Nn >> 4;
  if (wave < tiles_n) {
    const int col0 = wave << 4;
    const int half = lane >> 4;
    const int lid  = lane & 15;
    const int nchunks = K >> 5;

    const __bf16* arow = &Abf[lid * K];
    const __bf16* bpw  = Bp + (size_t)(wave * nchunks) * 32 * 16;

    v8f acc = {};
    for (int c = 0; c < nchunks; ++c) {
      const int kk = c << 5;
      const v8bf a0 = *(const v8bf*)(arow + kk + half * 8);
      const v8bf a1 = *(const v8bf*)(arow + kk + 16 + half * 8);
      const v16bf af = __builtin_shufflevector(a0, a1,
          0, 1, 2, 3, 4, 5, 6, 7, 8, 9, 10, 11, 12, 13, 14, 15);
      const v16bf bf = *(const v16bf*)(bpw + ((size_t)c * 32 + lane) * 16);
      acc = __builtin_amdgcn_wmma_f32_16x16x32_bf16(false, af, false, bf,
                                                    (short)0, acc, false, false);
    }

    const float bv = bias ? bias[col0 + lid] : 0.0f;
#pragma unroll
    for (int r = 0; r < 8; ++r) {
      const int m = row0 + r + 8 * half;     // C layout: v[r] -> row r (+8 hi half)
      C[(size_t)m * Nn + (col0 + lid)] = acc[r] + bv;
    }
  }
}

// ---------------------------------------------------------------------------
// Per-node dual dot products: hs[i] = h[i].a, hd[i] = h[i].b  (one wave/node)
// ---------------------------------------------------------------------------
__global__ void k_dot2(const float* __restrict__ h, const float* __restrict__ va,
                       const float* __restrict__ vb, float* __restrict__ outa,
                       float* __restrict__ outb, int n) {
  const int wid  = blockIdx.x * (blockDim.x >> 5) + (threadIdx.x >> 5);
  const int lane = threadIdx.x & 31;
  if (wid >= n) return;
  const float4 hv = *(const float4*)(h + (size_t)wid * DIM + lane * 4);
  const float4 a4 = *(const float4*)(va + lane * 4);
  const float4 b4 = *(const float4*)(vb + lane * 4);
  float sa = hv.x * a4.x + hv.y * a4.y + hv.z * a4.z + hv.w * a4.w;
  float sb = hv.x * b4.x + hv.y * b4.y + hv.z * b4.z + hv.w * b4.w;
#pragma unroll
  for (int off = 16; off; off >>= 1) {
    sa += __shfl_down(sa, off);
    sb += __shfl_down(sb, off);
  }
  if (lane == 0) { outa[wid] = sa; outb[wid] = sb; }
}

// ---------------------------------------------------------------------------
// Edge passes (E real edges + N self-loops appended)
// ---------------------------------------------------------------------------
__global__ void k_edge_max(const int* __restrict__ srcE, const int* __restrict__ dstE,
                           const float* __restrict__ hs, const float* __restrict__ hd,
                           float* __restrict__ smax, int ne, int n) {
  const int e = blockIdx.x * blockDim.x + threadIdx.x;
  if (e >= ne + n) return;
  int s, d;
  if (e < ne) { s = srcE[e]; d = dstE[e]; } else { s = d = e - ne; }
  const float l = leakyf(hs[s] + hd[d], GAT_SLOPE_C);
  atomicMaxF(&smax[d], l);
}

__global__ void k_edge_exp(const int* __restrict__ srcE, const int* __restrict__ dstE,
                           const float* __restrict__ hs, const float* __restrict__ hd,
                           const float* __restrict__ smax, float* __restrict__ ssum,
                           float* __restrict__ ew, int ne, int n) {
  const int e = blockIdx.x * blockDim.x + threadIdx.x;
  if (e >= ne + n) return;
  int s, d;
  if (e < ne) { s = srcE[e]; d = dstE[e]; } else { s = d = e - ne; }
  const float l  = leakyf(hs[s] + hd[d], GAT_SLOPE_C);
  const float ev = __expf(l - smax[d]);
  ew[e] = ev;
  atomicAdd(&ssum[d], ev);
}

// one wave per edge: agg[dst] += (ew/ssum[dst]) * h[src]
__global__ void k_edge_agg(const int* __restrict__ srcE, const int* __restrict__ dstE,
                           const float* __restrict__ ew, const float* __restrict__ ssum,
                           const float* __restrict__ h, float* __restrict__ agg,
                           int ne, int n) {
  const int wid  = blockIdx.x * (blockDim.x >> 5) + (threadIdx.x >> 5);
  const int lane = threadIdx.x & 31;
  if (wid >= ne + n) return;
  int s, d;
  if (wid < ne) { s = srcE[wid]; d = dstE[wid]; } else { s = d = wid - ne; }
  const float coef = ew[wid] / ssum[d];
  const float4 hv = *(const float4*)(h + (size_t)s * DIM + lane * 4);
  float* o = agg + (size_t)d * DIM + lane * 4;
  atomicAdd(o + 0, coef * hv.x);
  atomicAdd(o + 1, coef * hv.y);
  atomicAdd(o + 2, coef * hv.z);
  atomicAdd(o + 3, coef * hv.w);
}

__global__ void k_bias_leaky(float* __restrict__ p, const float* __restrict__ bias,
                             int total, int cols, float slope) {
  const int i = blockIdx.x * blockDim.x + threadIdx.x;
  if (i >= total) return;
  const float v = p[i] + bias[i % cols];
  p[i] = leakyf(v, slope);
}

// ---------------------------------------------------------------------------
// Global attention pooling
// ---------------------------------------------------------------------------
__global__ void k_attn_score(const float* __restrict__ h, const float* __restrict__ w,
                             const float* __restrict__ battn, float* __restrict__ sc,
                             float* __restrict__ gmax, int n) {
  const int wid  = blockIdx.x * (blockDim.x >> 5) + (threadIdx.x >> 5);
  const int lane = threadIdx.x & 31;
  if (wid >= n) return;
  const float4 hv = *(const float4*)(h + (size_t)wid * DIM + lane * 4);
  const float4 wv = *(const float4*)(w + lane * 4);
  float s = hv.x * wv.x + hv.y * wv.y + hv.z * wv.z + hv.w * wv.w;
#pragma unroll
  for (int off = 16; off; off >>= 1) s += __shfl_down(s, off);
  if (lane == 0) {
    s += battn[0];
    sc[wid] = s;
    atomicMaxF(gmax, s);
  }
}

__global__ void k_attn_norm(const float* __restrict__ sc, const float* __restrict__ gmax,
                            float* __restrict__ attnw, float* __restrict__ gsum, int n) {
  const int i = blockIdx.x * blockDim.x + threadIdx.x;
  float ev = 0.0f;
  if (i < n) {
    ev = __expf(sc[i] - gmax[0]);
    attnw[i] = ev;
  }
  float t = ev;
#pragma unroll
  for (int off = 16; off; off >>= 1) t += __shfl_down(t, off);
  if ((threadIdx.x & 31) == 0) atomicAdd(gsum, t);
}

__global__ void k_pool(const float* __restrict__ h, const float* __restrict__ attnw,
                       const float* __restrict__ gsum, const int* __restrict__ batch,
                       float* __restrict__ pooled, int n) {
  const int wid  = blockIdx.x * (blockDim.x >> 5) + (threadIdx.x >> 5);
  const int lane = threadIdx.x & 31;
  if (wid >= n) return;
  const float coef = attnw[wid] / gsum[0];
  const int g = batch[wid];
  const float4 hv = *(const float4*)(h + (size_t)wid * DIM + lane * 4);
  float* o = pooled + (size_t)g * DIM + lane * 4;
  atomicAdd(o + 0, coef * hv.x);
  atomicAdd(o + 1, coef * hv.y);
  atomicAdd(o + 2, coef * hv.z);
  atomicAdd(o + 3, coef * hv.w);
}

// ---------------------------------------------------------------------------
// BatchNorm (batch statistics over G rows, per column)
// ---------------------------------------------------------------------------
__global__ void k_bn_stats(const float* __restrict__ z, float* __restrict__ mean,
                           float* __restrict__ var, int rows, int cols) {
  const int c    = blockIdx.x * (blockDim.x >> 5) + (threadIdx.x >> 5);
  const int lane = threadIdx.x & 31;
  if (c >= cols) return;
  float s = 0.0f, sq = 0.0f;
  for (int r = lane; r < rows; r += 32) {
    const float v = z[(size_t)r * cols + c];
    s += v; sq += v * v;
  }
#pragma unroll
  for (int off = 16; off; off >>= 1) {
    s  += __shfl_down(s, off);
    sq += __shfl_down(sq, off);
  }
  if (lane == 0) {
    const float m = s / rows;
    mean[c] = m;
    var[c]  = sq / rows - m * m;   // biased variance (jnp.var default)
  }
}

__global__ void k_bn_apply(const float* __restrict__ z, float* __restrict__ out,
                           const float* __restrict__ mean, const float* __restrict__ var,
                           const float* __restrict__ g, const float* __restrict__ b,
                           int total, int cols) {
  const int i = blockIdx.x * blockDim.x + threadIdx.x;
  if (i >= total) return;
  const int c = i % cols;
  const float v = (z[i] - mean[c]) * rsqrtf(var[c] + BN_EPS_C) * g[c] + b[c];
  out[i] = leakyf(v, ACT_SLOPE_C);
}

// ---------------------------------------------------------------------------
// Host launcher
// ---------------------------------------------------------------------------
extern "C" void kernel_launch(void* const* d_in, const int* in_sizes, int n_in,
                              void* d_out, int out_size, void* d_ws, size_t ws_size,
                              hipStream_t stream) {
  const float* x      = (const float*)d_in[0];
  const int*   eidx   = (const int*)  d_in[1];
  const int*   batch  = (const int*)  d_in[2];
  const float* W1     = (const float*)d_in[3];
  const float* a_src1 = (const float*)d_in[4];
  const float* a_dst1 = (const float*)d_in[5];
  const float* b1     = (const float*)d_in[6];
  const float* W2     = (const float*)d_in[7];
  const float* a_src2 = (const float*)d_in[8];
  const float* a_dst2 = (const float*)d_in[9];
  const float* b2     = (const float*)d_in[10];
  const float* w_attn = (const float*)d_in[11];
  const float* b_attn = (const float*)d_in[12];
  const float* fc_w1  = (const float*)d_in[13];
  const float* fc_b1  = (const float*)d_in[14];
  const float* bn_g   = (const float*)d_in[15];
  const float* bn_b   = (const float*)d_in[16];
  const float* fc_w2  = (const float*)d_in[17];
  const float* fc_b2  = (const float*)d_in[18];

  float* ws     = (float*)d_ws;
  float* hbuf   = ws;                                   // N*D
  float* agg    = hbuf   + (size_t)N_NODES * DIM;       // N*D
  float* hs     = agg    + (size_t)N_NODES * DIM;       // N  (also attn scores)
  float* hd     = hs     + N_NODES;                     // N  (also attn weights)
  float* smax   = hd     + N_NODES;                     // N
  float* ssum   = smax   + N_NODES;                     // N
  float* ew     = ssum   + N_NODES;                     // E+N
  float* pooled = ew     + ETOTAL;                      // G*D
  float* z1     = pooled + (size_t)NGRAPH * DIM;        // G*C
  float* z2     = z1     + (size_t)NGRAPH * CHALF;      // G*C
  float* bnmean = z2     + (size_t)NGRAPH * CHALF;      // C
  float* bnvar  = bnmean + CHALF;                       // C
  float* gred   = bnvar  + CHALF;                       // [0]=gmax, [1]=gsum
  __bf16* bpack = (__bf16*)(((uintptr_t)(gred + 2) + 31u) & ~(uintptr_t)31u); // 32KB max

  const int* srcE = eidx;
  const int* dstE = eidx + N_EDGES;

  const dim3 blk(256);
  const dim3 gNodeWave((N_NODES + 7) / 8);
  const dim3 gEdgeThr((ETOTAL + 255) / 256);
  const dim3 gEdgeWave((ETOTAL + 7) / 8);
  const dim3 gNodeElem((N_NODES * DIM + 255) / 256);

  // ---- GAT layer 1 ----
  k_pack_b<<<dim3(4), blk, 0, stream>>>(W1, bpack, DIM, DIM);              // 32 waves
  k_wmma_gemm<<<dim3(N_NODES / 16), blk, 0, stream>>>(x, bpack, (const float*)nullptr,
                                                      hbuf, N_NODES, DIM, DIM);
  k_dot2<<<gNodeWave, blk, 0, stream>>>(hbuf, a_src1, a_dst1, hs, hd, N_NODES);
  k_fill<<<dim3(1024), blk, 0, stream>>>(smax, -INFINITY, (size_t)N_NODES);
  k_fill<<<dim3(1024), blk, 0, stream>>>(ssum, 0.0f, (size_t)N_NODES);
  k_fill<<<dim3(4096), blk, 0, stream>>>(agg, 0.0f, (size_t)N_NODES * DIM);
  k_edge_max<<<gEdgeThr, blk, 0, stream>>>(srcE, dstE, hs, hd, smax, N_EDGES, N_NODES);
  k_edge_exp<<<gEdgeThr, blk, 0, stream>>>(srcE, dstE, hs, hd, smax, ssum, ew, N_EDGES, N_NODES);
  k_edge_agg<<<gEdgeWave, blk, 0, stream>>>(srcE, dstE, ew, ssum, hbuf, agg, N_EDGES, N_NODES);
  k_bias_leaky<<<gNodeElem, blk, 0, stream>>>(agg, b1, N_NODES * DIM, DIM, ACT_SLOPE_C);

  // ---- GAT layer 2 ----
  k_pack_b<<<dim3(4), blk, 0, stream>>>(W2, bpack, DIM, DIM);
  k_wmma_gemm<<<dim3(N_NODES / 16), blk, 0, stream>>>(agg, bpack, (const float*)nullptr,
                                                      hbuf, N_NODES, DIM, DIM);
  k_dot2<<<gNodeWave, blk, 0, stream>>>(hbuf, a_src2, a_dst2, hs, hd, N_NODES);
  k_fill<<<dim3(1024), blk, 0, stream>>>(smax, -INFINITY, (size_t)N_NODES);
  k_fill<<<dim3(1024), blk, 0, stream>>>(ssum, 0.0f, (size_t)N_NODES);
  k_fill<<<dim3(4096), blk, 0, stream>>>(agg, 0.0f, (size_t)N_NODES * DIM);
  k_edge_max<<<gEdgeThr, blk, 0, stream>>>(srcE, dstE, hs, hd, smax, N_EDGES, N_NODES);
  k_edge_exp<<<gEdgeThr, blk, 0, stream>>>(srcE, dstE, hs, hd, smax, ssum, ew, N_EDGES, N_NODES);
  k_edge_agg<<<gEdgeWave, blk, 0, stream>>>(srcE, dstE, ew, ssum, hbuf, agg, N_EDGES, N_NODES);
  k_bias_leaky<<<gNodeElem, blk, 0, stream>>>(agg, b2, N_NODES * DIM, DIM, ACT_SLOPE_C);
  // `agg` now holds the final node features h.

  // ---- Global attention pooling ----
  k_fill<<<dim3(1), dim3(32), 0, stream>>>(gred, -INFINITY, (size_t)1);
  k_fill<<<dim3(1), dim3(32), 0, stream>>>(gred + 1, 0.0f, (size_t)1);
  k_fill<<<dim3(512), blk, 0, stream>>>(pooled, 0.0f, (size_t)NGRAPH * DIM);
  k_attn_score<<<gNodeWave, blk, 0, stream>>>(agg, w_attn, b_attn, hs, gred, N_NODES);
  k_attn_norm<<<dim3((N_NODES + 255) / 256), blk, 0, stream>>>(hs, gred, hd, gred + 1, N_NODES);
  k_pool<<<gNodeWave, blk, 0, stream>>>(agg, hd, gred + 1, batch, pooled, N_NODES);

  // ---- FC head: Linear -> BN -> LeakyReLU -> Linear ----
  k_pack_b<<<dim3(2), blk, 0, stream>>>(fc_w1, bpack, DIM, CHALF);         // 16 waves
  k_wmma_gemm<<<dim3(NGRAPH / 16), blk, 0, stream>>>(pooled, bpack, fc_b1, z1,
                                                     NGRAPH, DIM, CHALF);
  k_bn_stats<<<dim3(CHALF / 8), blk, 0, stream>>>(z1, bnmean, bnvar, NGRAPH, CHALF);
  k_bn_apply<<<dim3((NGRAPH * CHALF + 255) / 256), blk, 0, stream>>>(
      z1, z2, bnmean, bnvar, bn_g, bn_b, NGRAPH * CHALF, CHALF);
  k_pack_b<<<dim3(2), blk, 0, stream>>>(fc_w2, bpack, CHALF, DIM);         // 16 waves
  k_wmma_gemm<<<dim3(NGRAPH / 16), blk, 0, stream>>>(z2, bpack, fc_b2, (float*)d_out,
                                                     NGRAPH, CHALF, DIM);
}
